// Suspend_55825984913518
// MI455X (gfx1250) — matrix operand order repaired and checked by reading, not compiled, not run on hardware
//
#include <hip/hip_runtime.h>

// ---------------------------------------------------------------------------
// Problem: S=128, B=16, D=H=512, O=256.
// Algebraic collapse: watch/sus carries are uniform along their leading S axis
// (zero init + [B,H]-broadcast updates), so the scan is a [16,512] recurrence
// and outs[t,s,b,o] = out_t[b,o] for all s.
// MI455X cost model: recurrence ~2.4 GMAC (latency-bound, 1 persistent WGP of
// bf16 WMMA, weights bf16-packed + L2-resident); output write 536 MB ->
// ~23 us floor at 23.3 TB/s, handled by a streaming broadcast kernel.
// CDNA5 paths used: v_wmma_f32_16x16x32_bf16, Tensor Data Mover double-buffered
// prefetch of x[t+1] (tensor_load_to_lds + s_wait_tensorcnt), wave32 layouts.
// ---------------------------------------------------------------------------

typedef __attribute__((ext_vector_type(16))) __bf16 v16bf;
typedef __attribute__((ext_vector_type(8)))  float  v8f;
typedef __attribute__((ext_vector_type(4)))  unsigned int v4u;
typedef __attribute__((ext_vector_type(8)))  int v8i;
typedef __attribute__((ext_vector_type(4)))  int v4i;

union ABFrag { v16bf v; uint4 u[2]; };

namespace {
constexpr int S_LEN = 128;
constexpr int B_DIM = 16;
constexpr int D_DIM = 512;
constexpr int H_DIM = 512;
constexpr int O_DIM = 256;
constexpr int KT_H  = D_DIM / 32;   // 16 k-tiles (K=512, 32 per WMMA)
constexpr int NT_H  = H_DIM / 16;   // 32 n-tiles
constexpr int NT_O  = O_DIM / 16;   // 16 n-tiles
constexpr int XT_ELEMS = B_DIM * D_DIM;  // 8192 f32 per timestep (32 KB)
}

// ---------------------------------------------------------------------------
// Weight repack: W[K][N] f32 -> bf16 WMMA B-tiles.
// Tile (kt,nt) = 512 bf16 (32x16). Per ISA B layout: lanes 0-15 hold K={0..7,16..23}
// of column n=lane (k&31 = 8h + j + 8*(j>=8), h = lane>>4, 2 per VGPR); lanes 16-31 hold K={8..15,24..31} of
// column n=lane-16. Linear index: (kt*(N/16)+nt)*512 + lane*16 + j.
// ---------------------------------------------------------------------------
__global__ __launch_bounds__(256) void prep_w(const float* __restrict__ src,
                                              unsigned short* __restrict__ dst,
                                              int N, int total) {
  int i = blockIdx.x * blockDim.x + threadIdx.x;
  if (i >= total) return;
  const int tile = i >> 9;
  const int r    = i & 511;
  const int lane = r >> 4;
  const int j    = r & 15;
  const int ntiles = N >> 4;
  const int kt = tile / ntiles;
  const int nt = tile - kt * ntiles;
  const int k  = kt * 32 + ((j >> 3) << 4) + ((lane >> 4) << 3) + (j & 7);
  const int n  = nt * 16 + (lane & 15);
  union { __bf16 h; unsigned short u; } cv;
  cv.h = (__bf16)src[k * N + n];
  dst[i] = cv.u;
}

// ---------------------------------------------------------------------------
// TDM: async-DMA one contiguous 32 KB timestep of x (8192 f32, modeled as a
// 2D tensor: dim0=8192 elems of 4B, 1 row) from global memory into LDS.
// D# per CDNA5 ISA:
//   group0: [1:0]=count=1, [63:32]=lds_addr, [120:64]=global_addr, [127:126]=2
//   group1: [17:16]=data_size(2 -> 4B), [79:48]=tensor_dim0=8192,
//           [111:80]=tensor_dim1=1, [127:112]=tile_dim0=8192,
//           [207:160]=tensor_dim0_stride=8192; rest 0
//   groups 2/3 (+ trailing group on this toolchain's 6-arg builtin): zero,
//   tensor <= 2D.
// Tracked by TENSORcnt (per-wave): issuer waits s_wait_tensorcnt then the
// workgroup barrier publishes the LDS data.
// ---------------------------------------------------------------------------
__device__ __forceinline__ void tdm_load_x(const float* __restrict__ gsrc,
                                           unsigned int lds_off) {
  const unsigned long long ga = (unsigned long long)(uintptr_t)gsrc;
  v4u g0;
  g0[0] = 1u;                                            // count=1
  g0[1] = lds_off;                                       // lds_addr (bytes)
  g0[2] = (unsigned int)(ga & 0xFFFFFFFFu);              // global_addr[31:0]
  g0[3] = (unsigned int)((ga >> 32) & 0x01FFFFFFu)       // global_addr[56:32]
        | (2u << 30);                                    // type=2 ("image")
  v8i g1;
  g1[0] = (int)(2u << 16);                               // data_size=4B
  g1[1] = (int)((XT_ELEMS & 0xFFFF) << 16);              // tensor_dim0 lo16
  g1[2] = (int)((XT_ELEMS >> 16) | (1u << 16));          // dim0 hi16 | tensor_dim1=1
  g1[3] = (int)((XT_ELEMS & 0xFFFF) << 16);              // tile_dim0=8192
  g1[4] = 0;                                             // tile_dim1/2 unused
  g1[5] = XT_ELEMS;                                      // tensor_dim0_stride lo32
  g1[6] = 0;
  g1[7] = 0;
  v4i g2 = {0, 0, 0, 0};
  v4i g3 = {0, 0, 0, 0};
  v8i g4 = {0, 0, 0, 0, 0, 0, 0, 0};
  __builtin_amdgcn_tensor_load_to_lds(g0, g1, g2, g3, g4, 0);
}

// ---------------------------------------------------------------------------
// Multiply the staged 16x512 bf16 A (LDS) by NT consecutive 32x16 B tiles per
// k-step, accumulating f32. A fragment per lane = 32B = 2 x ds_read_b128.
// B fragment per lane = 2 x global_load_b128 (L2-resident packed weights).
// ---------------------------------------------------------------------------
template<int NT>
__device__ __forceinline__ void mm_tiles(const uint4* __restrict__ aStage,
                                         const uint4* __restrict__ wTiles,
                                         int ntBase, int nTilesTotal, int lane,
                                         v8f acc[NT]) {
  for (int kt = 0; kt < KT_H; ++kt) {
    ABFrag a;
    const uint4* ap = aStage + kt * 64 + lane * 2;
    a.u[0] = ap[0];
    a.u[1] = ap[1];
#pragma unroll
    for (int nt = 0; nt < NT; ++nt) {
      ABFrag b;
      const uint4* bp = wTiles + (size_t)(kt * nTilesTotal + ntBase + nt) * 64 + lane * 2;
      b.u[0] = bp[0];
      b.u[1] = bp[1];
      acc[nt] = __builtin_amdgcn_wmma_f32_16x16x32_bf16(
          false, a.v, false, b.v, (short)0, acc[nt], false, false);
    }
  }
}

// ---------------------------------------------------------------------------
// Stage a register tile (C/D layout: row = r + 8*(lane>=16),
// col = base + (lane&15) + 16*nt) into the LDS A-fragment layout:
// tile kt, lane la holds 16 bf16: la<16 -> K = kt*32 + {0..7,16..23},
// la>=16 -> K = kt*32 + {8..15,24..31}. Branchless index math.
// ---------------------------------------------------------------------------
__device__ __forceinline__ void stage_a(__bf16* __restrict__ dst,
                                        const float v[4][8], int lane, int base) {
  const int hi  = lane >> 4;
  const int c16 = lane & 15;
#pragma unroll
  for (int nt = 0; nt < 4; ++nt) {
    const int k  = base + c16 + nt * 16;
    const int kl = k & 31;
    const int j  = (kl & 7) | (((kl >> 4) & 1) << 3);
    const int laBase = ((kl >> 3) & 1) << 4;
    const int tileOff = (k >> 5) * 512;
#pragma unroll
    for (int r = 0; r < 8; ++r) {
      const int row = r + hi * 8;
      dst[tileOff + (row + laBase) * 16 + j] = (__bf16)v[nt][r];
    }
  }
}

// ---------------------------------------------------------------------------
// Persistent single-workgroup recurrence. 256 threads = 8 wave32.
// Wave w owns columns [64w, 64w+64) of all [16,512] tensors (registers),
// and columns [32w, 32w+32) of the [16,256] output.
// x[t+1] is TDM-prefetched into an LDS double buffer during step t.
// ---------------------------------------------------------------------------
__global__ __launch_bounds__(256, 1) void suspend_scan_kernel(
    const float* __restrict__ x,
    const float* __restrict__ attn_b, const float* __restrict__ in_b,
    const float* __restrict__ ins_b,  const float* __restrict__ sus_b,
    const float* __restrict__ out_b,
    const uint4* __restrict__ wAttn, const uint4* __restrict__ wIn,
    const uint4* __restrict__ wIns,  const uint4* __restrict__ wSus,
    const uint4* __restrict__ wOut,
    float* __restrict__ outbuf, float* __restrict__ watchbuf,
    float* __restrict__ susbuf) {
  __shared__ float ldsX[2][XT_ELEMS];  // 64KB: TDM double buffer for x[t]
  __shared__ uint4 ldsA[KT_H * 64];    // 16KB: A-stage for read / r
  __shared__ uint4 ldsW[KT_H * 64];    // 16KB: A-stage for watch (persistent)
  __shared__ float redA[16][8];
  __shared__ float redB[16][8];

  const int tid  = threadIdx.x;
  const int wv   = tid >> 5;
  const int lane = tid & 31;
  const int hi   = lane >> 4;
  const int c16  = lane & 15;
  const int base = wv * 64;

  // watch_0 = 0 -> zero its A-stage
  for (int i = tid; i < KT_H * 64; i += 256) ldsW[i] = make_uint4(0u, 0u, 0u, 0u);

  // Kick the TDM on x[0] while we set up biases.
  if (wv == 0)
    tdm_load_x(x, (unsigned int)(uintptr_t)&ldsX[0][0]);

  // Per-lane bias values (columns are fixed for the whole scan).
  float abv[4], ibv[4], sbv[4], obv[2];
#pragma unroll
  for (int nt = 0; nt < 4; ++nt) {
    const int col = base + c16 + nt * 16;
    abv[nt] = attn_b[col];
    ibv[nt] = in_b[col] + ins_b[col];
    sbv[nt] = sus_b[col];
  }
#pragma unroll
  for (int nt = 0; nt < 2; ++nt) obv[nt] = out_b[wv * 32 + c16 + nt * 16];

  float sreg[4][8];
  float wreg[4][8];
#pragma unroll
  for (int nt = 0; nt < 4; ++nt)
#pragma unroll
    for (int r = 0; r < 8; ++r) { sreg[nt][r] = 0.f; wreg[nt][r] = 0.f; }

  if (wv == 0) __builtin_amdgcn_s_wait_tensorcnt(0);
  __syncthreads();   // publishes ldsW zeros + x[0]

  for (int t = 0; t < S_LEN; ++t) {
    // Async-DMA next timestep into the other buffer (overlaps whole step).
    if (t + 1 < S_LEN && wv == 0)
      tdm_load_x(x + (size_t)(t + 1) * XT_ELEMS,
                 (unsigned int)(uintptr_t)&ldsX[(t + 1) & 1][0]);

    // ---- read x[t] from LDS into registers (C/D layout), stage as A ----
    const float* xs = &ldsX[t & 1][0];
    float xreg[4][8];
#pragma unroll
    for (int nt = 0; nt < 4; ++nt) {
      const int col = base + c16 + nt * 16;
#pragma unroll
      for (int r = 0; r < 8; ++r)
        xreg[nt][r] = xs[(r + hi * 8) * D_DIM + col];
    }
    stage_a((__bf16*)ldsA, xreg, lane, base);
    __syncthreads();

    // ---- matmul 1: logits = read @ attn_w + attn_b ----
    v8f acc1[4];
#pragma unroll
    for (int nt = 0; nt < 4; ++nt) acc1[nt] = (v8f){0.f,0.f,0.f,0.f,0.f,0.f,0.f,0.f};
    mm_tiles<4>(ldsA, wAttn, wv * 4, NT_H, lane, acc1);
#pragma unroll
    for (int nt = 0; nt < 4; ++nt)
#pragma unroll
      for (int r = 0; r < 8; ++r) acc1[nt][r] += abv[nt];

    // ---- row softmax over H=512 (intra-half shfl + cross-wave LDS) ----
    float pmax[8];
#pragma unroll
    for (int r = 0; r < 8; ++r)
      pmax[r] = fmaxf(fmaxf(acc1[0][r], acc1[1][r]), fmaxf(acc1[2][r], acc1[3][r]));
#pragma unroll
    for (int mk = 1; mk < 16; mk <<= 1)
#pragma unroll
      for (int r = 0; r < 8; ++r) pmax[r] = fmaxf(pmax[r], __shfl_xor(pmax[r], mk, 32));
    if (c16 == 0) {
#pragma unroll
      for (int r = 0; r < 8; ++r) redA[r + hi * 8][wv] = pmax[r];
    }
    __syncthreads();
    float rmax[8];
#pragma unroll
    for (int r = 0; r < 8; ++r) {
      float m = redA[r + hi * 8][0];
      for (int q = 1; q < 8; ++q) m = fmaxf(m, redA[r + hi * 8][q]);
      rmax[r] = m;
    }
    float psum[8];
#pragma unroll
    for (int r = 0; r < 8; ++r) psum[r] = 0.f;
#pragma unroll
    for (int nt = 0; nt < 4; ++nt)
#pragma unroll
      for (int r = 0; r < 8; ++r) {
        const float e = __expf(acc1[nt][r] - rmax[r]);
        acc1[nt][r] = e;
        psum[r] += e;
      }
#pragma unroll
    for (int mk = 1; mk < 16; mk <<= 1)
#pragma unroll
      for (int r = 0; r < 8; ++r) psum[r] += __shfl_xor(psum[r], mk, 32);
    if (c16 == 0) {
#pragma unroll
      for (int r = 0; r < 8; ++r) redB[r + hi * 8][wv] = psum[r];
    }
    __syncthreads();

    // ---- r = softmax * read ; stage as A ----
    float rreg[4][8];
#pragma unroll
    for (int r = 0; r < 8; ++r) {
      float s = redB[r + hi * 8][0];
      for (int q = 1; q < 8; ++q) s += redB[r + hi * 8][q];
      const float inv = 1.f / s;
#pragma unroll
      for (int nt = 0; nt < 4; ++nt)
        rreg[nt][r] = acc1[nt][r] * inv * xreg[nt][r];
    }
    stage_a((__bf16*)ldsA, rreg, lane, base);
    __syncthreads();

    // ---- matmul 2+3: watch = tanh(r@in_w + watch_old@ins_w + in_b + ins_b)
    v8f accP[4], accI[4];
#pragma unroll
    for (int nt = 0; nt < 4; ++nt) {
      accP[nt] = (v8f){0.f,0.f,0.f,0.f,0.f,0.f,0.f,0.f};
      accI[nt] = (v8f){0.f,0.f,0.f,0.f,0.f,0.f,0.f,0.f};
    }
    mm_tiles<4>(ldsA, wIn,  wv * 4, NT_H, lane, accP);
    mm_tiles<4>(ldsW, wIns, wv * 4, NT_H, lane, accI);
#pragma unroll
    for (int nt = 0; nt < 4; ++nt)
#pragma unroll
      for (int r = 0; r < 8; ++r)
        wreg[nt][r] = tanhf(accP[nt][r] + accI[nt][r] + ibv[nt]);

    __syncthreads();                 // everyone done reading watch_old A-stage
    stage_a((__bf16*)ldsW, wreg, lane, base);
    __syncthreads();

    // ---- matmul 4: sus = tanh(watch@sus_w + sus_b + sus) ----
    v8f accS[4];
#pragma unroll
    for (int nt = 0; nt < 4; ++nt) accS[nt] = (v8f){0.f,0.f,0.f,0.f,0.f,0.f,0.f,0.f};
    mm_tiles<4>(ldsW, wSus, wv * 4, NT_H, lane, accS);
#pragma unroll
    for (int nt = 0; nt < 4; ++nt)
#pragma unroll
      for (int r = 0; r < 8; ++r)
        sreg[nt][r] = tanhf(accS[nt][r] + sbv[nt] + sreg[nt][r]);

    // ---- matmul 5: out_t = watch@out_w + out_b -> compact buffer ----
    v8f accO[2];
#pragma unroll
    for (int nt = 0; nt < 2; ++nt) accO[nt] = (v8f){0.f,0.f,0.f,0.f,0.f,0.f,0.f,0.f};
    mm_tiles<2>(ldsW, wOut, wv * 2, NT_O, lane, accO);
    float* ot = outbuf + (size_t)t * (B_DIM * O_DIM);
#pragma unroll
    for (int nt = 0; nt < 2; ++nt) {
      const int col = wv * 32 + c16 + nt * 16;
#pragma unroll
      for (int r = 0; r < 8; ++r)
        ot[(r + hi * 8) * O_DIM + col] = accO[nt][r] + obv[nt];
    }

    // Retire this step's TDM prefetch, then publish the buffer to all waves.
    if (wv == 0) __builtin_amdgcn_s_wait_tensorcnt(0);
    __syncthreads();
  }

  // export final states (compact [16,512])
#pragma unroll
  for (int nt = 0; nt < 4; ++nt) {
    const int col = base + c16 + nt * 16;
#pragma unroll
    for (int r = 0; r < 8; ++r) {
      const int row = r + hi * 8;
      watchbuf[row * H_DIM + col] = wreg[nt][r];
      susbuf[row * H_DIM + col]   = sreg[nt][r];
    }
  }
}

// ---------------------------------------------------------------------------
// Streaming broadcast: the 536 MB of output. Reads stay in L2 (2 MB + 64 KB).
// d_out = [outs S*S*B*O][watch S*B*H][sus S*B*H], all f32, float4 stores.
// ---------------------------------------------------------------------------
__global__ __launch_bounds__(256) void bcast_kernel(
    const float4* __restrict__ outbuf, const float4* __restrict__ watchbuf,
    const float4* __restrict__ susbuf, float4* __restrict__ dst) {
  const size_t TOT = 33554432ull;  // S*S*B*O/4
  const size_t WS4 = 262144ull;    // S*B*H/4
  size_t i = (size_t)blockIdx.x * blockDim.x + threadIdx.x;
  if (i >= TOT + 2 * WS4) return;
  float4 v;
  if (i < TOT) {
    const size_t t = i >> 17;          // / (S * B*O/4) = /131072
    const size_t inner = i & 1023;     // % (B*O/4)
    v = outbuf[(t << 10) + inner];
  } else if (i < TOT + WS4) {
    v = watchbuf[(i - TOT) & 2047];    // % (B*H/4)
  } else {
    v = susbuf[(i - TOT - WS4) & 2047];
  }
  dst[i] = v;
}

// ---------------------------------------------------------------------------
extern "C" void kernel_launch(void* const* d_in, const int* in_sizes, int n_in,
                              void* d_out, int out_size, void* d_ws, size_t ws_size,
                              hipStream_t stream) {
  const float* x      = (const float*)d_in[0];
  const float* attn_w = (const float*)d_in[1];
  const float* attn_b = (const float*)d_in[2];
  const float* in_w   = (const float*)d_in[3];
  const float* in_b   = (const float*)d_in[4];
  const float* ins_w  = (const float*)d_in[5];
  const float* ins_b  = (const float*)d_in[6];
  const float* sus_w  = (const float*)d_in[7];
  const float* sus_b  = (const float*)d_in[8];
  const float* out_w  = (const float*)d_in[9];
  const float* out_b  = (const float*)d_in[10];

  char* ws = (char*)d_ws;
  const size_t WH  = 512ull * 512 * 2;   // bf16 H-weight: 512 KB
  const size_t WO  = 512ull * 256 * 2;   // bf16 out_w: 256 KB
  unsigned short* wbAttn = (unsigned short*)(ws);
  unsigned short* wbIn   = (unsigned short*)(ws + WH);
  unsigned short* wbIns  = (unsigned short*)(ws + 2 * WH);
  unsigned short* wbSus  = (unsigned short*)(ws + 3 * WH);
  unsigned short* wbOut  = (unsigned short*)(ws + 4 * WH);
  float* outbuf   = (float*)(ws + 4 * WH + WO);                       // 2 MB
  float* watchbuf = (float*)(ws + 4 * WH + WO + 2097152ull);          // 32 KB
  float* susbuf   = (float*)(ws + 4 * WH + WO + 2097152ull + 32768ull);

  // 1) repack weights to bf16 WMMA B-tile layout
  prep_w<<<(512 * 512 + 255) / 256, 256, 0, stream>>>(attn_w, wbAttn, 512, 512 * 512);
  prep_w<<<(512 * 512 + 255) / 256, 256, 0, stream>>>(in_w,  wbIn,  512, 512 * 512);
  prep_w<<<(512 * 512 + 255) / 256, 256, 0, stream>>>(ins_w, wbIns, 512, 512 * 512);
  prep_w<<<(512 * 512 + 255) / 256, 256, 0, stream>>>(sus_w, wbSus, 512, 512 * 512);
  prep_w<<<(512 * 256 + 255) / 256, 256, 0, stream>>>(out_w, wbOut, 256, 512 * 256);

  // 2) sequential recurrence on one persistent workgroup (8 wave32, WMMA bf16,
  //    TDM double-buffered x prefetch)
  suspend_scan_kernel<<<1, 256, 0, stream>>>(
      x, attn_b, in_b, ins_b, sus_b, out_b,
      (const uint4*)wbAttn, (const uint4*)wbIn, (const uint4*)wbIns,
      (const uint4*)wbSus, (const uint4*)wbOut,
      outbuf, watchbuf, susbuf);

  // 3) bandwidth-bound broadcast of the replicated outputs (34,078,720 float4)
  bcast_kernel<<<133120, 256, 0, stream>>>(
      (const float4*)outbuf, (const float4*)watchbuf, (const float4*)susbuf,
      (float4*)d_out);
}
